// SA_One_Head_16612933501467
// MI455X (gfx1250) — matrix operand order repaired and checked by reading, not compiled, not run on hardware
//
#include <hip/hip_runtime.h>
#include <stdint.h>

#define HEAD  128
#define EMBED 1024
#define TLEN  4096

typedef __attribute__((ext_vector_type(16))) __bf16 v16bf;
typedef __attribute__((ext_vector_type(8)))  float  v8f;

union FragB16 {
  v16bf v;
  uint4 q[2];
  unsigned short u[16];
};

__device__ __forceinline__ unsigned short f2bf(float f) {
  union { float f; unsigned u; } c;
  c.f = f;
  unsigned r = c.u + 0x7FFFu + ((c.u >> 16) & 1u);  // round-to-nearest-even
  return (unsigned short)(r >> 16);
}

__device__ __forceinline__ unsigned pack2bf(float lo, float hi) {
  return (unsigned)f2bf(lo) | ((unsigned)f2bf(hi) << 16);
}

// ---------------------------------------------------------------------------
// Kernel 0: repack Wq/Wk/Wv (f32 [C=1024][H=128]) into bf16 B-fragment order.
//   lane L: column n = L&15, K half = (L>=16)?16:0, 16 contiguous K values.
// wfrag: [mat(3)][htile(8)][cstep(32)][lane(32)][16] ushort
// ---------------------------------------------------------------------------
__global__ __launch_bounds__(256) void wmma_reorder_w(
    const float* __restrict__ Wq, const float* __restrict__ Wk,
    const float* __restrict__ Wv, unsigned short* __restrict__ wfrag) {
  int gtid = blockIdx.x * 256 + threadIdx.x;
  int lane = gtid & 31;
  int frag = gtid >> 5;
  if (frag >= 768) return;
  int cstep = frag & 31;
  int htile = (frag >> 5) & 7;
  int mat = frag >> 8;
  const float* W = (mat == 0) ? Wq : (mat == 1) ? Wk : Wv;
  int h = htile * 16 + (lane & 15);
  int c0 = cstep * 32 + ((lane >= 16) ? 16 : 0);
  FragB16 o;
#pragma unroll
  for (int j = 0; j < 16; ++j) o.u[j] = f2bf(W[(size_t)(c0 + j) * HEAD + h]);
  uint4* dst = (uint4*)(wfrag + ((size_t)frag * 32 + lane) * 16);
  dst[0] = o.q[0];
  dst[1] = o.q[1];
}

// ---------------------------------------------------------------------------
// Kernel 1: QKV projection, software-pipelined.
// Block = 256 threads (8 waves); block owns 16 rows of x; wave w owns h-tile w.
// Q gets scale = C^-0.5 * log2(e) folded in.  V stored transposed Vt[b][h][t].
// ---------------------------------------------------------------------------
#define XSTRIDE 1032  // 1024 + 8 pad (ushorts) -> conflict-free b128 LDS reads

__global__ __launch_bounds__(256) void qkv_proj(
    const float* __restrict__ x, const unsigned short* __restrict__ wfrag,
    unsigned short* __restrict__ Qbf, unsigned short* __restrict__ Kbf,
    unsigned short* __restrict__ Vt) {
  __shared__ unsigned short xt[16 * XSTRIDE];
  __shared__ unsigned short vstage[8 * 256];
  const int tid = threadIdx.x;
  const int lane = tid & 31;
  const int wave = tid >> 5;
  const int rowBase = blockIdx.x * 16;  // flattened B*T row
  const float* xb = x + (size_t)rowBase * EMBED;
  // cooperative f32 -> bf16 staging of the 16x1024 x tile (packed b64 stores)
#pragma unroll 4
  for (int i = tid; i < 16 * 256; i += 256) {
    int r = i >> 8;
    int c4 = (i & 255) * 4;
    float4 f = *(const float4*)(xb + (size_t)r * EMBED + c4);
    uint2 w;
    w.x = pack2bf(f.x, f.y);
    w.y = pack2bf(f.z, f.w);
    *(uint2*)&xt[r * XSTRIDE + c4] = w;
  }
  __syncthreads();

  const int htile = wave;
  const int row = lane & 15;
  const int agrp = (lane >= 16) ? 8 : 0;  // D row half; 2*agrp = A's contiguous K half
  v8f zero = {};
  v8f accQ = zero, accK = zero, accV = zero;
  const unsigned short* wq = wfrag + (size_t)htile * 32 * 512 + (size_t)lane * 16;

  FragB16 a[2], bq[2], bk[2], bv[2];
  auto load_frags = [&](int cs, int buf) {
    const unsigned short* ap = &xt[row * XSTRIDE + cs * 32 + 2 * agrp];
    a[buf].q[0] = *(const uint4*)ap;
    a[buf].q[1] = *(const uint4*)(ap + 8);
    const unsigned short* wp = wq + (size_t)cs * 512;
    bq[buf].q[0] = *(const uint4*)(wp);
    bq[buf].q[1] = *(const uint4*)(wp + 8);
    bk[buf].q[0] = *(const uint4*)(wp + 131072);
    bk[buf].q[1] = *(const uint4*)(wp + 131072 + 8);
    bv[buf].q[0] = *(const uint4*)(wp + 262144);
    bv[buf].q[1] = *(const uint4*)(wp + 262144 + 8);
  };

  load_frags(0, 0);
#pragma unroll 2
  for (int cs = 0; cs < 32; ++cs) {
    const int cur = cs & 1, nxt = cur ^ 1;
    if (cs + 1 < 32) load_frags(cs + 1, nxt);  // in flight during WMMAs below
    accQ = __builtin_amdgcn_wmma_f32_16x16x32_bf16(false, a[cur].v, false, bq[cur].v, (short)0, accQ, false, false);
    accK = __builtin_amdgcn_wmma_f32_16x16x32_bf16(false, a[cur].v, false, bk[cur].v, (short)0, accK, false, false);
    accV = __builtin_amdgcn_wmma_f32_16x16x32_bf16(false, a[cur].v, false, bv[cur].v, (short)0, accV, false, false);
  }

  const float QS = 0.04508422002778011f;  // 1024^-0.5 * log2(e)
  const int col = htile * 16 + row;
#pragma unroll
  for (int r = 0; r < 8; ++r) {
    size_t t = (size_t)(rowBase + r + agrp);
    Qbf[t * HEAD + col] = f2bf(accQ[r] * QS);
    Kbf[t * HEAD + col] = f2bf(accK[r]);
  }
  // transpose V tile through LDS (same-wave DS ops are in order)
  unsigned short* vs = &vstage[wave * 256];
#pragma unroll
  for (int r = 0; r < 8; ++r) vs[row * 16 + r + agrp] = f2bf(accV[r]);
  const int b = rowBase >> 12;
  const int t0 = rowBase & (TLEN - 1);
  const int hl = lane >> 1, half = (lane & 1) * 8;
  uint4 val = *(const uint4*)&vs[hl * 16 + half];
  *(uint4*)(Vt + ((size_t)b * HEAD + htile * 16 + hl) * TLEN + t0 + half) = val;
}

// ---------------------------------------------------------------------------
// Kernel 2: flash attention.  One wave per 16-query block, key steps of 32.
// K fragments batched before the S WMMAs; V fragments issued before softmax
// so their latency hides behind exp2/shuffle VALU work.
// ---------------------------------------------------------------------------
#define PASTRIDE 48  // ushorts; multiple of 8 so b128 reads stay aligned

__global__ __launch_bounds__(128) void flash_attn(
    const unsigned short* __restrict__ Qbf, const unsigned short* __restrict__ Kbf,
    const unsigned short* __restrict__ Vt, float* __restrict__ out) {
  __shared__ unsigned short pA[4][16 * PASTRIDE];
  const int lane = threadIdx.x & 31;
  const int wave = threadIdx.x >> 5;
  const int b = blockIdx.y;
  const int t0 = (blockIdx.x * 4 + wave) * 16;
  const unsigned short* Qb = Qbf + (size_t)b * TLEN * HEAD;
  const unsigned short* Kb = Kbf + (size_t)b * TLEN * HEAD;
  const unsigned short* Vb = Vt + (size_t)b * HEAD * TLEN;
  const int row = lane & 15;
  const int agrp = (lane >= 16) ? 8 : 0;   // accumulator row half
  const int bgrp = (lane >= 16) ? 16 : 0;  // B contiguous half

  FragB16 aq[4];
#pragma unroll
  for (int cs = 0; cs < 4; ++cs) {
    const unsigned short* p = Qb + (size_t)(t0 + row) * HEAD + cs * 32 + bgrp;
    aq[cs].q[0] = *(const uint4*)p;
    aq[cs].q[1] = *(const uint4*)(p + 8);
  }

  float m[8], l[8];
  v8f zero = {};
  v8f acc[8];
#pragma unroll
  for (int r = 0; r < 8; ++r) { m[r] = -1e30f; l[r] = 0.0f; }
#pragma unroll
  for (int ht = 0; ht < 8; ++ht) acc[ht] = zero;
  unsigned short* myPA = pA[wave];

  for (int k0 = 0; k0 < TLEN; k0 += 32) {
    // warm WGP$ two steps ahead: 32 K rows (one per lane), 128 Vt rows
    if (k0 + 64 < TLEN) {
      const unsigned short* kpf = Kb + (size_t)(k0 + 64 + lane) * HEAD;
      __builtin_prefetch(kpf, 0, 3);
      __builtin_prefetch(kpf + 64, 0, 3);
#pragma unroll
      for (int j = 0; j < 4; ++j)
        __builtin_prefetch(Vb + (size_t)(j * 32 + lane) * TLEN + k0 + 64, 0, 3);
    }
    // --- batch all K fragments for this step (16 b128 loads in flight) ---
    const unsigned short* kp0 = Kb + (size_t)(k0 + row) * HEAD + bgrp;
    const unsigned short* kp1 = Kb + (size_t)(k0 + 16 + row) * HEAD + bgrp;
    FragB16 kf[8];
#pragma unroll
    for (int cs = 0; cs < 4; ++cs) {
      kf[cs * 2 + 0].q[0] = *(const uint4*)(kp0 + cs * 32);
      kf[cs * 2 + 0].q[1] = *(const uint4*)(kp0 + cs * 32 + 8);
      kf[cs * 2 + 1].q[0] = *(const uint4*)(kp1 + cs * 32);
      kf[cs * 2 + 1].q[1] = *(const uint4*)(kp1 + cs * 32 + 8);
    }
    // --- S = Q K^T (log2 domain) ---
    v8f s0 = zero, s1 = zero;
#pragma unroll
    for (int cs = 0; cs < 4; ++cs) {
      s0 = __builtin_amdgcn_wmma_f32_16x16x32_bf16(false, aq[cs].v, false, kf[cs * 2 + 0].v, (short)0, s0, false, false);
      s1 = __builtin_amdgcn_wmma_f32_16x16x32_bf16(false, aq[cs].v, false, kf[cs * 2 + 1].v, (short)0, s1, false, false);
    }
    // --- issue V fragment loads now; consumed after softmax ---
    FragB16 vf[8];
#pragma unroll
    for (int ht = 0; ht < 8; ++ht) {
      const unsigned short* vp = Vb + (size_t)(ht * 16 + row) * TLEN + k0 + bgrp;
      vf[ht].q[0] = *(const uint4*)vp;
      vf[ht].q[1] = *(const uint4*)(vp + 8);
    }
    // --- online softmax ---
    float tm[8];
#pragma unroll
    for (int r = 0; r < 8; ++r) tm[r] = fmaxf(s0[r], s1[r]);
#pragma unroll
    for (int off = 1; off <= 8; off <<= 1)
#pragma unroll
      for (int r = 0; r < 8; ++r) tm[r] = fmaxf(tm[r], __shfl_xor(tm[r], off, 32));
    float alpha[8], rs[8];
#pragma unroll
    for (int r = 0; r < 8; ++r) {
      float mn = fmaxf(m[r], tm[r]);
      alpha[r] = exp2f(m[r] - mn);
      m[r] = mn;
      float p0 = exp2f(s0[r] - mn);
      float p1 = exp2f(s1[r] - mn);
      myPA[(r + agrp) * PASTRIDE + row] = f2bf(p0);
      myPA[(r + agrp) * PASTRIDE + 16 + row] = f2bf(p1);
      rs[r] = p0 + p1;
    }
#pragma unroll
    for (int off = 1; off <= 8; off <<= 1)
#pragma unroll
      for (int r = 0; r < 8; ++r) rs[r] += __shfl_xor(rs[r], off, 32);
#pragma unroll
    for (int r = 0; r < 8; ++r) l[r] = l[r] * alpha[r] + rs[r];
#pragma unroll
    for (int ht = 0; ht < 8; ++ht)
#pragma unroll
      for (int r = 0; r < 8; ++r) acc[ht][r] *= alpha[r];
    // --- P (C layout) -> A fragment via LDS, then acc += P V ---
    FragB16 ap;
    const unsigned short* pp = &myPA[row * PASTRIDE + bgrp];
    ap.q[0] = *(const uint4*)pp;
    ap.q[1] = *(const uint4*)(pp + 8);
#pragma unroll
    for (int ht = 0; ht < 8; ++ht)
      acc[ht] = __builtin_amdgcn_wmma_f32_16x16x32_bf16(false, ap.v, false, vf[ht].v, (short)0, acc[ht], false, false);
  }

  float rl[8];
#pragma unroll
  for (int r = 0; r < 8; ++r) rl[r] = 1.0f / l[r];
  float* ob = out + ((size_t)b * TLEN + t0) * HEAD;
#pragma unroll
  for (int ht = 0; ht < 8; ++ht)
#pragma unroll
    for (int r = 0; r < 8; ++r)
      ob[(size_t)(r + agrp) * HEAD + ht * 16 + row] = acc[ht][r] * rl[r];
}

// ---------------------------------------------------------------------------
extern "C" void kernel_launch(void* const* d_in, const int* in_sizes, int n_in,
                              void* d_out, int out_size, void* d_ws, size_t ws_size,
                              hipStream_t stream) {
  (void)in_sizes; (void)n_in; (void)out_size; (void)ws_size;
  const float* x  = (const float*)d_in[0];
  const float* Wk = (const float*)d_in[1];
  const float* Wq = (const float*)d_in[2];
  const float* Wv = (const float*)d_in[3];
  float* out = (float*)d_out;

  // workspace layout (bytes): wfrag 786432 | Qbf 4M | Kbf 4M | Vt 4M
  char* ws = (char*)d_ws;
  unsigned short* wfrag = (unsigned short*)(ws);
  unsigned short* Qbf   = (unsigned short*)(ws + 786432);
  unsigned short* Kbf   = (unsigned short*)(ws + 786432 + 4194304);
  unsigned short* Vt    = (unsigned short*)(ws + 786432 + 2 * 4194304);

  hipLaunchKernelGGL(wmma_reorder_w, dim3(96), dim3(256), 0, stream, Wq, Wk, Wv, wfrag);
  hipLaunchKernelGGL(qkv_proj, dim3(1024), dim3(256), 0, stream, x, wfrag, Qbf, Kbf, Vt);
  hipLaunchKernelGGL(flash_attn, dim3(64, 4), dim3(128), 0, stream, Qbf, Kbf, Vt, out);
}
